// Encoder_31078383354578
// MI455X (gfx1250) — matrix operand order repaired and checked by reading, not compiled, not run on hardware
//
#include <hip/hip_runtime.h>
#include <hip/hip_bf16.h>

typedef __attribute__((ext_vector_type(16))) __bf16 v16bf;
typedef __attribute__((ext_vector_type(8)))  float  v8f;
typedef int v4i_ __attribute__((vector_size(16)));

union Frag16 { uint4 q[2]; v16bf v; };

#define B_  64
#define T_  256
#define I_  512
#define H_  800
#define G4H 3200
#define WTP 832   // per-gate padded column count for WT (13 tiles of 64)

#if defined(__gfx1250__) && \
    __has_builtin(__builtin_amdgcn_global_load_async_to_lds_b128) && \
    __has_builtin(__builtin_amdgcn_s_wait_asynccnt)
#define USE_ASYNC_LDS 1
#else
#define USE_ASYNC_LDS 0
#endif

__device__ __forceinline__ unsigned short f2bf(float f) {
  unsigned int u = __float_as_uint(f);
  u += 0x7FFFu + ((u >> 16) & 1u);
  return (unsigned short)(u >> 16);
}
__device__ __forceinline__ unsigned int pack2bf(float a, float b) {
  // lowers to v_cvt_pk_bf16_f32
  union { __hip_bfloat162 h2; unsigned int u; } cv;
  cv.h2 = __float22bfloat162_rn(make_float2(a, b));
  return cv.u;
}
__device__ __forceinline__ float sigmoidf_(float x) {
  return 1.0f / (1.0f + __expf(-x));
}
__device__ __forceinline__ float tanhf_(float x) {
  float e = __expf(-2.0f * x);
  return (1.0f - e) / (1.0f + e);
}

// ---------------- one-time weight transpose + bf16 convert ----------------
__global__ void transpose_W_kernel(const float* __restrict__ W,
                                   unsigned short* __restrict__ WT, int n) {
  int i = blockIdx.x * blockDim.x + threadIdx.x;
  if (i < n) {
    int k = i / G4H;
    int c = i - k * G4H;
    int g = c / H_;
    int cl = c - g * H_;
    WT[(unsigned)(g * WTP + cl) * (unsigned)I_ + (unsigned)k] = f2bf(W[i]);
  }
}
__global__ void transpose_U_kernel(const float* __restrict__ U,
                                   unsigned short* __restrict__ UT, int n) {
  int i = blockIdx.x * blockDim.x + threadIdx.x;
  if (i < n) {
    int k = i / G4H;
    int c = i - k * G4H;
    UT[(unsigned)c * (unsigned)H_ + (unsigned)k] = f2bf(U[i]);
  }
}

// ---------------- zero init h (bf16 phase 0) + c (f32) ----------------
__global__ void zero_state_kernel(unsigned short* __restrict__ hbf0,
                                  float* __restrict__ c, int n) {
  int i = blockIdx.x * blockDim.x + threadIdx.x;
  if (i < n) { hbf0[i] = 0; c[i] = 0.0f; }
}

// ---------------- input projection GEMM (masked x @ W + b) ----------------
// grid: (256 M-tiles, 13 N-tiles-of-64-within-gate, dir*4+gate)
// block: 128 threads = 4 waves; each wave computes a 64x16 column strip.
__global__ __launch_bounds__(128)
void proj_gemm_kernel(const float* __restrict__ x,
                      const float* __restrict__ mask_f,
                      const float* __restrict__ mask_b,
                      const unsigned short* __restrict__ WT_f,
                      const unsigned short* __restrict__ WT_b,
                      const float* __restrict__ bias_f,
                      const float* __restrict__ bias_b,
                      float* __restrict__ proj_f,
                      float* __restrict__ proj_b) {
  __shared__ __align__(16) unsigned short sA[64 * 40];     // 64 rows x 32 k
  __shared__ __align__(16) unsigned short sB[2][64 * 40];  // double-buffered B

  const int tid   = threadIdx.x;
  const int wave  = tid >> 5;
  const int lane  = tid & 31;
  const int l15   = lane & 15;
  const int khalf = lane >> 4;

  const int bm  = blockIdx.x;
  const int bn  = blockIdx.y;
  const int g   = blockIdx.z & 3;
  const int dir = blockIdx.z >> 2;

  const float* mask         = dir ? mask_b : mask_f;
  const unsigned short* WT  = dir ? WT_b : WT_f;
  const float* bias         = dir ? bias_b : bias_f;
  float* proj               = dir ? proj_b : proj_f;

  const unsigned wtbase = (unsigned)(g * WTP + bn * 64) * (unsigned)I_;

  const int ncol = bn * 64 + wave * 16 + l15;   // column within gate
  const bool colok = (ncol < H_);
  const float bv = colok ? bias[g * H_ + ncol] : 0.0f;

  v8f acc[4];
#pragma unroll
  for (int m = 0; m < 4; ++m)
#pragma unroll
    for (int r = 0; r < 8; ++r) acc[m][r] = bv;

#if USE_ASYNC_LDS
  // issue B tile 0 (2 async b128 per wave)
  {
#pragma unroll
    for (int it = 0; it < 2; ++it) {
      int p = it * 128 + tid;
      int n = p >> 2;
      int q = p & 3;
      const unsigned short* src = WT + wtbase + (unsigned)n * (unsigned)I_ +
                                  (unsigned)(q * 8);
      unsigned short* dst = &sB[0][n * 40 + q * 8];
      __builtin_amdgcn_global_load_async_to_lds_b128(
          (v4i_*)(void*)src, (v4i_*)(void*)dst, 0, 0);
    }
  }
#endif

  for (int kk = 0; kk < I_ / 32; ++kk) {
    const int k0 = kk * 32;
    const int cur = kk & 1;

#if USE_ASYNC_LDS
    if (kk + 1 < I_ / 32) {  // issue next B tile into other buffer
#pragma unroll
      for (int it = 0; it < 2; ++it) {
        int p = it * 128 + tid;
        int n = p >> 2;
        int q = p & 3;
        const unsigned short* src = WT + wtbase + (unsigned)n * (unsigned)I_ +
                                    (unsigned)(k0 + 32 + q * 8);
        unsigned short* dst = &sB[1 - cur][n * 40 + q * 8];
        __builtin_amdgcn_global_load_async_to_lds_b128(
            (v4i_*)(void*)src, (v4i_*)(void*)dst, 0, 0);
      }
    }
#else
#pragma unroll
    for (int it = 0; it < 8; ++it) {
      int p  = it * 128 + tid;
      int n  = p & 63;
      int k2 = (p >> 6) * 2;
      *(unsigned int*)(&sB[cur][n * 40 + k2]) =
          *(const unsigned int*)(WT + wtbase + (unsigned)n * (unsigned)I_ +
                                 (unsigned)(k0 + k2));
    }
#endif
    // A tile: x * dropout-mask -> bf16, 4 k per thread-chunk (overlaps async)
#pragma unroll
    for (int it = 0; it < 4; ++it) {
      int p  = it * 128 + tid;   // 0..511
      int r  = p >> 3;           // row 0..63
      int k4 = (p & 7) * 4;      // 0,4,...,28
      int rowg = bm * 64 + r;
      int b  = rowg >> 8;
      int t  = rowg & 255;
      int tt = dir ? (T_ - 1 - t) : t;   // backward dir reads reversed time
      unsigned xoff = (unsigned)(b * T_ + tt) * (unsigned)I_ + (unsigned)(k0 + k4);
      unsigned moff = (unsigned)(b * 4 + g) * (unsigned)I_ + (unsigned)(k0 + k4);
      float a0 = x[xoff]      * mask[moff];
      float a1 = x[xoff + 1u] * mask[moff + 1u];
      float a2 = x[xoff + 2u] * mask[moff + 2u];
      float a3 = x[xoff + 3u] * mask[moff + 3u];
      uint2 pk;
      pk.x = pack2bf(a0, a1);
      pk.y = pack2bf(a2, a3);
      *(uint2*)(sA + r * 40 + k4) = pk;
    }
#if USE_ASYNC_LDS
    if (kk + 1 < I_ / 32) __builtin_amdgcn_s_wait_asynccnt(2);  // cur tile done
    else                  __builtin_amdgcn_s_wait_asynccnt(0);
#endif
    __syncthreads();

    Frag16 bf;
    const char* pb = (const char*)&sB[cur][0] + (wave * 16 + l15) * 80 + khalf * 16;
    bf.q[0] = *(const uint4*)(pb);
    bf.q[1] = *(const uint4*)(pb + 32);
#pragma unroll
    for (int m = 0; m < 4; ++m) {
      Frag16 af;
      const char* pa = (const char*)sA + (m * 16 + l15) * 80 + khalf * 16;
      af.q[0] = *(const uint4*)(pa);
      af.q[1] = *(const uint4*)(pa + 32);
      acc[m] = __builtin_amdgcn_wmma_f32_16x16x32_bf16(
          false, af.v, false, bf.v, (short)0, acc[m], false, false);
    }
    __syncthreads();
  }

  if (colok) {
#pragma unroll
    for (int m = 0; m < 4; ++m)
#pragma unroll
      for (int r = 0; r < 8; ++r) {
        unsigned row = (unsigned)(bm * 64 + m * 16 + r + khalf * 8);
        proj[row * (unsigned)G4H + (unsigned)(g * H_ + ncol)] = acc[m][r];
      }
  }
}

// ---------------- fused recurrent step: z = proj_t + h@U; gates; c,h ----
// grid: (50 unit-tiles of 16, 2 dirs); block 128 = 4 waves, wave w = gate w.
// Block owns all 4 gates of 16 hidden units -> pointwise update is local.
__global__ __launch_bounds__(128)
void lstm_step_kernel(const unsigned short* __restrict__ hbf_rd,
                      unsigned short* __restrict__ hbf_wr,
                      float* __restrict__ cbuf,
                      const unsigned short* __restrict__ UT_f,
                      const unsigned short* __restrict__ UT_b,
                      const float* __restrict__ proj_f,
                      const float* __restrict__ proj_b,
                      float* __restrict__ out,
                      int t, int writeFinal) {
  __shared__ __align__(16) unsigned char smem[2 * 10240];  // 20480 B
  // per buffer: sH (64x40 bf16, 5120B) then sU (64x40 bf16, 5120B)
  float* zb = (float*)smem;  // 64x68 f32 = 17408B, reused after the k-loop

  const int tid   = threadIdx.x;
  const int wave  = tid >> 5;     // gate index
  const int lane  = tid & 31;
  const int l15   = lane & 15;
  const int khalf = lane >> 4;

  const int dir = blockIdx.y;
  const int u0  = blockIdx.x * 16;

  const unsigned short* hin  = hbf_rd + dir * (B_ * H_);
  unsigned short*       hout = hbf_wr + dir * (B_ * H_);
  float*                cst  = cbuf   + dir * (B_ * H_);
  const unsigned short* UT   = dir ? UT_b : UT_f;
  const float*          proj = dir ? proj_b : proj_f;

  v8f acc[4];
#pragma unroll
  for (int m = 0; m < 4; ++m)
#pragma unroll
    for (int r = 0; r < 8; ++r) acc[m][r] = 0.0f;

#if USE_ASYNC_LDS
  // issue tile 0 into buffer 0 (4 async b128 per wave: 2 for sH, 2 for sU)
  {
    unsigned short* sH0 = (unsigned short*)smem;
    unsigned short* sU0 = (unsigned short*)(smem + 5120);
#pragma unroll
    for (int it = 0; it < 2; ++it) {
      int p = it * 128 + tid;
      int r = p >> 2;
      int q = p & 3;
      const unsigned short* srcH = hin + (unsigned)r * (unsigned)H_ + (unsigned)(q * 8);
      __builtin_amdgcn_global_load_async_to_lds_b128(
          (v4i_*)(void*)srcH, (v4i_*)(void*)(sH0 + r * 40 + q * 8), 0, 0);
    }
#pragma unroll
    for (int it = 0; it < 2; ++it) {
      int p = it * 128 + tid;
      int n = p >> 2;
      int q = p & 3;
      unsigned c = (unsigned)((n >> 4) * H_ + u0 + (n & 15));
      const unsigned short* srcU = UT + c * (unsigned)H_ + (unsigned)(q * 8);
      __builtin_amdgcn_global_load_async_to_lds_b128(
          (v4i_*)(void*)srcU, (v4i_*)(void*)(sU0 + n * 40 + q * 8), 0, 0);
    }
  }
#endif

  for (int kk = 0; kk < H_ / 32; ++kk) {
    const int k0 = kk * 32;
    const int cur = kk & 1;
    unsigned short* sH = (unsigned short*)(smem + cur * 10240);
    unsigned short* sU = (unsigned short*)(smem + cur * 10240 + 5120);

    // prefetch two tiles ahead (the async pipeline covers k+1)
    if (k0 + 64 < H_) {
      __builtin_prefetch(UT + (unsigned)(wave * H_ + u0 + l15) * (unsigned)H_ +
                             (unsigned)(k0 + 64), 0, 1);
    }

#if USE_ASYNC_LDS
    if (kk + 1 < H_ / 32) {  // issue next tile into the other buffer
      unsigned short* nH = (unsigned short*)(smem + (1 - cur) * 10240);
      unsigned short* nU = (unsigned short*)(smem + (1 - cur) * 10240 + 5120);
      const int kn = k0 + 32;
#pragma unroll
      for (int it = 0; it < 2; ++it) {
        int p = it * 128 + tid;
        int r = p >> 2;
        int q = p & 3;
        const unsigned short* srcH = hin + (unsigned)r * (unsigned)H_ +
                                     (unsigned)(kn + q * 8);
        __builtin_amdgcn_global_load_async_to_lds_b128(
            (v4i_*)(void*)srcH, (v4i_*)(void*)(nH + r * 40 + q * 8), 0, 0);
      }
#pragma unroll
      for (int it = 0; it < 2; ++it) {
        int p = it * 128 + tid;
        int n = p >> 2;
        int q = p & 3;
        unsigned c = (unsigned)((n >> 4) * H_ + u0 + (n & 15));
        const unsigned short* srcU = UT + c * (unsigned)H_ + (unsigned)(kn + q * 8);
        __builtin_amdgcn_global_load_async_to_lds_b128(
            (v4i_*)(void*)srcU, (v4i_*)(void*)(nU + n * 40 + q * 8), 0, 0);
      }
      __builtin_amdgcn_s_wait_asynccnt(4);  // oldest group (tile kk) complete
    } else {
      __builtin_amdgcn_s_wait_asynccnt(0);
    }
#else
#pragma unroll
    for (int it = 0; it < 8; ++it) {
      int p  = it * 128 + tid;
      int r  = p >> 4;
      int k2 = (p & 15) * 2;
      *(unsigned int*)(sH + r * 40 + k2) =
          *(const unsigned int*)(hin + (unsigned)r * (unsigned)H_ +
                                 (unsigned)(k0 + k2));
    }
#pragma unroll
    for (int it = 0; it < 8; ++it) {
      int p  = it * 128 + tid;
      int n  = p & 63;
      int k2 = (p >> 6) * 2;
      unsigned c = (unsigned)((n >> 4) * H_ + u0 + (n & 15));
      *(unsigned int*)(sU + n * 40 + k2) =
          *(const unsigned int*)(UT + c * (unsigned)H_ + (unsigned)(k0 + k2));
    }
#endif
    __syncthreads();

    Frag16 bf;
    const char* pb = (const char*)sU + (wave * 16 + l15) * 80 + khalf * 16;
    bf.q[0] = *(const uint4*)(pb);
    bf.q[1] = *(const uint4*)(pb + 32);
#pragma unroll
    for (int m = 0; m < 4; ++m) {
      Frag16 af;
      const char* pa = (const char*)sH + (m * 16 + l15) * 80 + khalf * 16;
      af.q[0] = *(const uint4*)(pa);
      af.q[1] = *(const uint4*)(pa + 32);
      acc[m] = __builtin_amdgcn_wmma_f32_16x16x32_bf16(
          false, af.v, false, bf.v, (short)0, acc[m], false, false);
    }
    __syncthreads();
  }

  // stage z into LDS so every thread can see all 4 gates of its units
#pragma unroll
  for (int m = 0; m < 4; ++m)
#pragma unroll
    for (int r = 0; r < 8; ++r) {
      int row = m * 16 + r + khalf * 8;
      zb[row * 68 + wave * 16 + l15] = acc[m][r];
    }
  __syncthreads();

  const unsigned HBASE = (unsigned)B_ * T_ * 1600u;
#pragma unroll
  for (int it = 0; it < 8; ++it) {
    int e  = it * 128 + tid;
    int b  = e >> 4;
    int ul = e & 15;
    int uu = u0 + ul;
    unsigned prow = (unsigned)(b * T_ + t) * (unsigned)G4H;
    float zi = zb[b * 68 + ul]      + proj[prow + (unsigned)(0 * H_ + uu)];
    float zf = zb[b * 68 + 16 + ul] + proj[prow + (unsigned)(1 * H_ + uu)];
    float zg = zb[b * 68 + 32 + ul] + proj[prow + (unsigned)(2 * H_ + uu)];
    float zo = zb[b * 68 + 48 + ul] + proj[prow + (unsigned)(3 * H_ + uu)];
    float ig = sigmoidf_(zi);
    float fg = sigmoidf_(zf);
    float gg = tanhf_(zg);
    float og = sigmoidf_(zo);
    unsigned soff = (unsigned)(b * H_ + uu);
    float cold = cst[soff];
    float cn = fg * cold + ig * gg;
    float hn = og * tanhf_(cn);
    cst[soff]  = cn;
    hout[soff] = f2bf(hn);
    out[(unsigned)(b * T_ + t) * 1600u + (unsigned)(dir * H_ + uu)] = hn;
    if (writeFinal) {
      out[HBASE + (unsigned)b * 1600u + (unsigned)(dir * H_ + uu)] = hn;
      out[HBASE + (unsigned)B_ * 1600u + (unsigned)b * 1600u +
          (unsigned)(dir * H_ + uu)] = cn;
    }
  }
}

extern "C" void kernel_launch(void* const* d_in, const int* in_sizes, int n_in,
                              void* d_out, int out_size, void* d_ws, size_t ws_size,
                              hipStream_t stream) {
  const float* x      = (const float*)d_in[0];
  const float* mask_f = (const float*)d_in[1];
  const float* mask_b = (const float*)d_in[2];
  const float* W_f    = (const float*)d_in[3];
  const float* U_f    = (const float*)d_in[4];
  const float* b_f    = (const float*)d_in[5];
  const float* W_b    = (const float*)d_in[6];
  const float* U_b    = (const float*)d_in[7];
  const float* b_b    = (const float*)d_in[8];
  float* out = (float*)d_out;

  char* ws = (char*)d_ws;
  size_t off = 0;
  auto carve = [&](size_t bytes) -> char* {
    char* p = ws + off;
    off += (bytes + 255) & ~(size_t)255;
    return p;
  };

  const size_t nW  = (size_t)I_ * G4H;        // 1,638,400 source elems
  const size_t nU  = (size_t)H_ * G4H;        // 2,560,000 source elems
  const size_t nWT = (size_t)4 * WTP * I_;    // 1,703,936 padded WT elems
  unsigned short* WT_f = (unsigned short*)carve(nWT * 2);
  unsigned short* WT_b = (unsigned short*)carve(nWT * 2);
  unsigned short* UT_f = (unsigned short*)carve(nU * 2);
  unsigned short* UT_b = (unsigned short*)carve(nU * 2);
  // h ping-pong: [phase][dir][B*H] bf16
  unsigned short* hbf   = (unsigned short*)carve((size_t)2 * 2 * B_ * H_ * 2);
  float*          cbuf  = (float*)carve((size_t)2 * B_ * H_ * 4);
  float*          projf = (float*)carve((size_t)B_ * T_ * G4H * 4);
  float*          projb = (float*)carve((size_t)B_ * T_ * G4H * 4);

  transpose_W_kernel<<<(int)((nW + 255) / 256), 256, 0, stream>>>(W_f, WT_f, (int)nW);
  transpose_W_kernel<<<(int)((nW + 255) / 256), 256, 0, stream>>>(W_b, WT_b, (int)nW);
  transpose_U_kernel<<<(int)((nU + 255) / 256), 256, 0, stream>>>(U_f, UT_f, (int)nU);
  transpose_U_kernel<<<(int)((nU + 255) / 256), 256, 0, stream>>>(U_b, UT_b, (int)nU);

  const int nS = 2 * B_ * H_;  // 102,400 (both dirs)
  zero_state_kernel<<<(nS + 255) / 256, 256, 0, stream>>>(hbf, cbuf, nS);

  proj_gemm_kernel<<<dim3(256, 13, 8), 128, 0, stream>>>(
      x, mask_f, mask_b, WT_f, WT_b, b_f, b_b, projf, projb);

  for (int t = 0; t < T_; ++t) {
    const unsigned short* hrd = hbf + (size_t)(t & 1) * nS;
    unsigned short*       hwr = hbf + (size_t)((t + 1) & 1) * nS;
    lstm_step_kernel<<<dim3(50, 2), 128, 0, stream>>>(
        hrd, hwr, cbuf, UT_f, UT_b, projf, projb, out, t, (t == T_ - 1) ? 1 : 0);
  }
}
